// MultiPolicyFedG_3307124818435
// MI455X (gfx1250) — matrix operand order repaired and checked by reading, not compiled, not run on hardware
//
#include <hip/hip_runtime.h>
#include <hip/hip_bf16.h>

// ---------------------------------------------------------------------------
// GINEConv x2 + policy head for MI455X (gfx1250, wave32, WMMA + TDM).
// Per conv:
//   1) zero aggr
//   2) edge kernel: one wave per edge, m = relu(x[src] + ea*We + be),
//      global_atomic_add_f32 into aggr[dst]            (bandwidth/atomic bound)
//   3) combine: Hh = f16(x + aggr)
//   4) pack W1,W2 (f32 128x128) into WMMA B-fragment lane layout (f16)
//   5) fused MLP:
//        - TDM (tensor_load_to_lds) stages both packed weight matrices into
//          LDS once per workgroup (s_wait_tensorcnt + barrier)
//        - per-wave 16-row tile: mid = relu(Hh@W1+b1) staged in wave-private
//          LDS (A-fragment layout), out = mid@W2+b2 (+relu)
//        - v_wmma_f32_16x16x32_f16, B operands from LDS (ds_load_b128)
//        - C tile bounced through LDS for coalesced global_store_b128
// ---------------------------------------------------------------------------

typedef __attribute__((ext_vector_type(16))) _Float16 v16h;
typedef __attribute__((ext_vector_type(8)))  _Float16 v8h;
typedef __attribute__((ext_vector_type(8)))  float    v8f;
typedef __attribute__((ext_vector_type(4)))  unsigned int u32x4;
typedef __attribute__((ext_vector_type(8)))  int      i32x8;
typedef __attribute__((ext_vector_type(4)))  int      i32x4;

union AFrag { v16h v; v8h h[2]; };

#define DH 128

#if __has_builtin(__builtin_amdgcn_tensor_load_to_lds) && \
    __has_builtin(__builtin_amdgcn_s_wait_tensorcnt)
#define USE_TDM 1
#else
#define USE_TDM 0
#endif

// ---------------- zero ----------------
__global__ __launch_bounds__(256) void zero_kernel(float* __restrict__ p, long n) {
  long i = (long)blockIdx.x * blockDim.x + threadIdx.x;
  if (i < n) p[i] = 0.0f;
}

// ---------------- edge message + scatter-add ----------------
// one wave (32 lanes) per edge; lane handles 4 contiguous channels
__global__ __launch_bounds__(256) void edge_kernel(
    const float* __restrict__ xin, const int* __restrict__ src,
    const int* __restrict__ dst, const float* __restrict__ eattr,
    const float* __restrict__ We, const float* __restrict__ be,
    float* __restrict__ aggr, int E_) {
  int gtid = blockIdx.x * blockDim.x + threadIdx.x;
  int eid  = gtid >> 5;
  int lane = gtid & 31;
  if (eid >= E_) return;
  int s = src[eid];
  int d = dst[eid];
  float ea = eattr[eid];
  int c = lane * 4;
  const float4 xv = *(const float4*)(xin + (size_t)s * DH + c);
  const float4 wv = *(const float4*)(We + c);
  const float4 bv = *(const float4*)(be + c);
  float m0 = fmaxf(xv.x + ea * wv.x + bv.x, 0.0f);
  float m1 = fmaxf(xv.y + ea * wv.y + bv.y, 0.0f);
  float m2 = fmaxf(xv.z + ea * wv.z + bv.z, 0.0f);
  float m3 = fmaxf(xv.w + ea * wv.w + bv.w, 0.0f);
  float* ap = aggr + (size_t)d * DH + c;
  atomicAdd(ap + 0, m0);
  atomicAdd(ap + 1, m1);
  atomicAdd(ap + 2, m2);
  atomicAdd(ap + 3, m3);
}

// ---------------- H = f16(x + aggr) ----------------
__global__ __launch_bounds__(256) void combine_kernel(
    const float* __restrict__ xin, const float* __restrict__ aggr,
    _Float16* __restrict__ Hh, long n) {
  long i = (long)blockIdx.x * blockDim.x + threadIdx.x;
  if (i < n) Hh[i] = (_Float16)(xin[i] + aggr[i]);
}

// ---------------- pack 128x128 f32 weight (K-major rows) into B fragments ---
// Fragment (nt,ks): 32 lanes x 16 halves contiguous.
//   lane -> col = nt*16 + (lane&15); kpart = lane>>4
//   j    -> K   = ks*32 + kpart*16 + j
__global__ __launch_bounds__(256) void pack_kernel(
    const float* __restrict__ W, _Float16* __restrict__ Wp) {
  int gid = blockIdx.x * blockDim.x + threadIdx.x;   // 0..16383
  if (gid >= 128 * 128) return;
  int j    = gid & 15;
  int lane = (gid >> 4) & 31;
  int fs   = gid >> 9;          // 0..31 = nt*4+ks
  int nt   = fs >> 2;
  int ks   = fs & 3;
  int col  = nt * 16 + (lane & 15);
  int krow = ks * 32 + (lane >> 4) * 16 + j;
  Wp[gid] = (_Float16)W[krow * DH + col];
}

#if USE_TDM
// Stage one contiguous block of `nelem` f16 elements from global into LDS.
// D# per CDNA5 ISA ch.8: 2D tensor, single row tile, data_size=2B.
__device__ __forceinline__ void tdm_load_w(unsigned lds_byte_off,
                                           const _Float16* g, int nelem) {
  unsigned long long ga = (unsigned long long)(uintptr_t)g;
  u32x4 g0;
  g0[0] = 1u;                                        // count=1 (valid), user mode
  g0[1] = lds_byte_off;                              // lds_addr (bytes)
  g0[2] = (unsigned)(ga & 0xFFFFFFFFull);            // global_addr[31:0]
  g0[3] = (unsigned)((ga >> 32) & 0x1FFFFFFull)      // global_addr[56:32]
          | (2u << 30);                              // type = 2 ("image")
  i32x8 g1;
  g1[0] = (1 << 16);                                 // data_size=1 -> 2 bytes
  g1[1] = (nelem & 0xFFFF) << 16;                    // tensor_dim0 [15:0]
  g1[2] = ((nelem >> 16) & 0xFFFF) | (1 << 16);      // tensor_dim0 hi | tensor_dim1=1
  g1[3] = (nelem & 0xFFFF) << 16;                    // tile_dim0
  g1[4] = 1;                                         // tile_dim1=1
  g1[5] = nelem;                                     // tensor_dim0_stride lo32
  g1[6] = 0;
  g1[7] = 0;
  i32x4 z4 = {0, 0, 0, 0};                           // groups 2/3 unused (<=2D)
  i32x8 z8 = {0, 0, 0, 0, 0, 0, 0, 0};
  // 6-arg form (clang-23 / amdgpu-toolchain): (g0, g1, g2, g3, g4, cpol)
  __builtin_amdgcn_tensor_load_to_lds(g0, g1, z4, z4, z8, 0);
}
#endif

// ---------------- fused 2-layer MLP, WMMA f16 -> f32 ----------------
__global__ __launch_bounds__(128) void mlp_kernel(
    const _Float16* __restrict__ Hh,
    const _Float16* __restrict__ W1p, const float* __restrict__ b1,
    const _Float16* __restrict__ W2p, const float* __restrict__ b2,
    float* __restrict__ out, int nMblk, int relu_out) {
  __shared__ _Float16 sW1[128 * 128];        // 32KB packed B fragments, layer 1
  __shared__ _Float16 sW2[128 * 128];        // 32KB packed B fragments, layer 2
  __shared__ _Float16 smid[4][16 * DH];      // 16KB per-wave mid tile (A layout)
  __shared__ float    sout[4][16 * DH];      // 32KB per-wave C staging

  int wave = threadIdx.x >> 5;
  int lane = threadIdx.x & 31;
  int mblk = blockIdx.x * 4 + wave;

  // ---- stage both weight matrices into LDS once per workgroup ----
#if USE_TDM
  if (wave == 0) {                           // TDM issue is per-wave; EXEC ignored
    tdm_load_w((unsigned)(uintptr_t)(void*)sW1, W1p, 128 * 128);
    tdm_load_w((unsigned)(uintptr_t)(void*)sW2, W2p, 128 * 128);
    __builtin_amdgcn_s_wait_tensorcnt(0);
  }
#else
  for (int i = threadIdx.x; i < (128 * 128) / 8; i += 128) {
    ((uint4*)sW1)[i] = ((const uint4*)W1p)[i];
    ((uint4*)sW2)[i] = ((const uint4*)W2p)[i];
  }
#endif
  __syncthreads();                           // all waves see staged weights

  if (mblk >= nMblk) return;                 // tail waves exit after barrier;
                                             // wave-uniform -> EXEC all-1s below
  int row  = lane & 15;                      // A: M = lane&15 ; C: N = lane&15
  int half = lane >> 4;                      // K-half select (A), M-half (C)
  size_t rowbase = ((size_t)mblk * 16 + row) * DH;

  // ---- A fragments for GEMM1 from global f16 (two 16B loads per K-step) ----
  AFrag a[4];
#pragma unroll
  for (int ks = 0; ks < 4; ++ks) {
    int k1 = ks * 32 + half * 16;
    a[ks].h[0] = *(const v8h*)(Hh + rowbase + k1);
    a[ks].h[1] = *(const v8h*)(Hh + rowbase + k1 + 8);
  }

  _Float16* myMid = smid[wave];
  float*    myOut = sout[wave];

  // ---- GEMM1: mid = relu(H@W1+b1), store transposed into LDS as f16 ----
#pragma unroll
  for (int nt = 0; nt < 8; ++nt) {
    v8f acc = {};
#pragma unroll
    for (int ks = 0; ks < 4; ++ks) {
      v16h bf = *(const v16h*)(sW1 + ((nt * 4 + ks) * 32 + lane) * 16);
      acc = __builtin_amdgcn_wmma_f32_16x16x32_f16(
          false, a[ks].v, false, bf, (short)0, acc, false, false);
    }
    int col = nt * 16 + row;
    float bias = b1[col];
#pragma unroll
    for (int r = 0; r < 8; ++r) {
      float v = fmaxf(acc[r] + bias, 0.0f);
      myMid[(half * 8 + r) * DH + col] = (_Float16)v;
    }
  }

  // wave-private LDS region; ds ops are in-order per wave, wait to be safe
  asm volatile("s_wait_dscnt 0" ::: "memory");

  // ---- A fragments for GEMM2 from LDS ----
  AFrag a2[4];
#pragma unroll
  for (int ks = 0; ks < 4; ++ks) {
    int k1 = ks * 32 + half * 16;
    a2[ks].h[0] = *(const v8h*)(myMid + row * DH + k1);
    a2[ks].h[1] = *(const v8h*)(myMid + row * DH + k1 + 8);
  }

  // ---- GEMM2: out = mid@W2 + b2 (+relu) -> LDS staging ----
#pragma unroll
  for (int nt = 0; nt < 8; ++nt) {
    v8f acc = {};
#pragma unroll
    for (int ks = 0; ks < 4; ++ks) {
      v16h bf = *(const v16h*)(sW2 + ((nt * 4 + ks) * 32 + lane) * 16);
      acc = __builtin_amdgcn_wmma_f32_16x16x32_f16(
          false, a2[ks].v, false, bf, (short)0, acc, false, false);
    }
    int col = nt * 16 + row;
    float bias = b2[col];
#pragma unroll
    for (int r = 0; r < 8; ++r) {
      float v = acc[r] + bias;
      if (relu_out) v = fmaxf(v, 0.0f);
      myOut[(half * 8 + r) * DH + col] = v;   // banked scatter, cheap in LDS
    }
  }

  asm volatile("s_wait_dscnt 0" ::: "memory");

  // ---- coalesced writeback: lane -> (row = lane>>1, 64-col half-row) ----
  int orow = lane >> 1;
  int ocol = (lane & 1) * 64;
  const float4* sp = (const float4*)(myOut + orow * DH + ocol);
  float4* gp = (float4*)(out + ((size_t)mblk * 16 + orow) * DH + ocol);
#pragma unroll
  for (int c = 0; c < 4; ++c) gp[c] = sp[c];
}

// ---------------- policy head ----------------
__global__ __launch_bounds__(128) void head_kernel(
    const float* __restrict__ embs, const int* __restrict__ curr,
    const int* __restrict__ dest, const int* __restrict__ nbrs,
    const float* __restrict__ Wl1, const float* __restrict__ bl1,
    const float* __restrict__ Wl2, const float* __restrict__ bl2,
    float* __restrict__ out) {
  __shared__ float sc[DH], sd[DH], sn[16][DH], sh[16][DH];
  int t = threadIdx.x;                       // 0..127
  int ci = curr[0], di = dest[0];
  sc[t] = embs[(size_t)ci * DH + t];
  sd[t] = embs[(size_t)di * DH + t];
  for (int k = 0; k < 16; ++k) sn[k][t] = embs[(size_t)nbrs[k] * DH + t];
  __syncthreads();

  float base = bl1[t];
  for (int j = 0; j < DH; ++j) base += sc[j] * Wl1[j * DH + t];
  for (int j = 0; j < DH; ++j) base += sd[j] * Wl1[(DH + j) * DH + t];
  for (int k = 0; k < 16; ++k) {
    float acc = base;
    for (int j = 0; j < DH; ++j) acc += sn[k][j] * Wl1[(2 * DH + j) * DH + t];
    sh[k][t] = fmaxf(acc, 0.0f);
  }
  __syncthreads();

  int wv = t >> 5, ln = t & 31;              // 4 waves of 32 (wave32)
  for (int k = wv; k < 16; k += 4) {
    float p = 0.0f;
    for (int j = ln; j < DH; j += 32) p += sh[k][j] * Wl2[j];
    for (int off = 16; off > 0; off >>= 1) p += __shfl_down(p, off, 32);
    if (ln == 0) out[k] = p + bl2[0];
  }
}

// ---------------------------------------------------------------------------
extern "C" void kernel_launch(void* const* d_in, const int* in_sizes, int n_in,
                              void* d_out, int out_size, void* d_ws, size_t ws_size,
                              hipStream_t stream) {
  const float* x    = (const float*)d_in[0];
  const int*   ei   = (const int*)d_in[1];
  const int*   curr = (const int*)d_in[2];
  const int*   dest = (const int*)d_in[3];
  const int*   nbrs = (const int*)d_in[4];
  const float* eat  = (const float*)d_in[5];
  const float* We1  = (const float*)d_in[6];  const float* be1 = (const float*)d_in[7];
  const float* W11  = (const float*)d_in[8];  const float* b11 = (const float*)d_in[9];
  const float* W12  = (const float*)d_in[10]; const float* b12 = (const float*)d_in[11];
  const float* We2  = (const float*)d_in[12]; const float* be2 = (const float*)d_in[13];
  const float* W21  = (const float*)d_in[14]; const float* b21 = (const float*)d_in[15];
  const float* W22  = (const float*)d_in[16]; const float* b22 = (const float*)d_in[17];
  const float* Wl1  = (const float*)d_in[18]; const float* bl1 = (const float*)d_in[19];
  const float* Wl2  = (const float*)d_in[20]; const float* bl2 = (const float*)d_in[21];

  const int  n  = in_sizes[0] / DH;        // 100000
  const int  e  = in_sizes[1] / 2;         // 1600000
  const long NF = (long)n * DH;

  // workspace layout
  float*     aggr = (float*)d_ws;              // NF f32 (also final embeddings)
  float*     hbuf = aggr + NF;                 // NF f32 (conv1 output)
  _Float16*  Hh   = (_Float16*)(hbuf + NF);    // NF f16
  _Float16*  W1p  = Hh + NF;                   // 16384 f16
  _Float16*  W2p  = W1p + 128 * 128;           // 16384 f16

  const int zb    = (int)((NF + 255) / 256);
  const int eb    = (int)(((long)e * 32 + 255) / 256);
  const int nMblk = (n + 15) / 16;
  const int gb    = (nMblk + 3) / 4;

  // ---- conv1 ----
  zero_kernel<<<zb, 256, 0, stream>>>(aggr, NF);
  edge_kernel<<<eb, 256, 0, stream>>>(x, ei, ei + e, eat, We1, be1, aggr, e);
  combine_kernel<<<zb, 256, 0, stream>>>(x, aggr, Hh, NF);
  pack_kernel<<<64, 256, 0, stream>>>(W11, W1p);
  pack_kernel<<<64, 256, 0, stream>>>(W12, W2p);
  mlp_kernel<<<gb, 128, 0, stream>>>(Hh, W1p, b11, W2p, b12, hbuf, nMblk, 1);

  // ---- conv2 ----
  zero_kernel<<<zb, 256, 0, stream>>>(aggr, NF);
  edge_kernel<<<eb, 256, 0, stream>>>(hbuf, ei, ei + e, eat, We2, be2, aggr, e);
  combine_kernel<<<zb, 256, 0, stream>>>(hbuf, aggr, Hh, NF);
  pack_kernel<<<64, 256, 0, stream>>>(W21, W1p);
  pack_kernel<<<64, 256, 0, stream>>>(W22, W2p);
  // reuse aggr as node-embedding output (no longer needed after combine)
  mlp_kernel<<<gb, 128, 0, stream>>>(Hh, W1p, b21, W2p, b22, aggr, nMblk, 0);

  // ---- head ----
  head_kernel<<<1, 128, 0, stream>>>(aggr, curr, dest, nbrs,
                                     Wl1, bl1, Wl2, bl2, (float*)d_out);
}
